// Attention_EV_29119878267501
// MI455X (gfx1250) — matrix operand order repaired and checked by reading, not compiled, not run on hardware
//
#include <hip/hip_runtime.h>

typedef __attribute__((ext_vector_type(16))) __bf16 v16bf;
typedef __attribute__((ext_vector_type(8)))  __bf16 v8bf;
typedef __attribute__((ext_vector_type(8)))  float  v8f;

#define NB   4096   // batch rows
#define INF  4096   // features
#define PE   256    // embed dim E
#define NCHUNK 8    // BN row chunks (512 rows each)

// ---------------- BN stage 1: partial column sums ----------------
__global__ void bn_partial(const float* __restrict__ x,
                           float* __restrict__ partS,
                           float* __restrict__ partS2) {
  int c = blockIdx.x * 256 + threadIdx.x;      // feature column
  int r0 = blockIdx.y * (NB / NCHUNK);
  float s = 0.f, s2 = 0.f;
#pragma unroll 4
  for (int r = 0; r < NB / NCHUNK; ++r) {
    float v = x[(size_t)(r0 + r) * INF + c];
    s += v; s2 += v * v;
  }
  partS [blockIdx.y * INF + c] = s;
  partS2[blockIdx.y * INF + c] = s2;
}

// ---------------- BN stage 2: scale/shift per feature ----------------
__global__ void bn_finalize(const float* __restrict__ partS,
                            const float* __restrict__ partS2,
                            const float* __restrict__ gamma,
                            const float* __restrict__ beta,
                            float* __restrict__ scale,
                            float* __restrict__ shift) {
  int c = blockIdx.x * 256 + threadIdx.x;
  float s = 0.f, s2 = 0.f;
#pragma unroll
  for (int j = 0; j < NCHUNK; ++j) { s += partS[j * INF + c]; s2 += partS2[j * INF + c]; }
  float mean = s * (1.f / NB);
  float var  = s2 * (1.f / NB) - mean * mean;
  float sc = rsqrtf(var + 1e-5f) * gamma[c];
  scale[c] = sc;
  shift[c] = beta[c] - mean * sc;
}

// ---------------- Weight prep: bf16 + WMMA B-lane swizzle ----------------
// B order used here (contiguous k, matching A): lane l holds N = l&15; element i holds
// K = i + 16*(l>>4). Stored per (proj, etile, ktile) tile, lane-major, so each
// lane's 16 bf16 values (32B) are contiguous.
__global__ void weight_swizzle(const float* __restrict__ WQ,
                               const float* __restrict__ WK,
                               const float* __restrict__ WV,
                               __bf16* __restrict__ Bswz) {
  int idx  = blockIdx.x * 256 + threadIdx.x;   // 0 .. 3*65536-1
  int proj = idx >> 16;
  int ek   = idx & 65535;
  int e = ek >> 8, k = ek & 255;
  const float* W = (proj == 0) ? WQ : ((proj == 1) ? WK : WV);
  float v = W[e * PE + k];                     // B[k][e] = W[e][k] (Q = xr @ W.T)
  int et = e >> 4, kt = k >> 5;
  int ln = 16 * ((k >> 4) & 1) + (e & 15);
  int i  = k & 15;
  Bswz[((((proj * 16 + et) * 8) + kt) * 32 + ln) * 16 + i] = (__bf16)v;
}

// ---------------- Fused attention: one workgroup per batch row ----------------
__launch_bounds__(256, 1)
__global__ void attn_fused(const float* __restrict__ x,
                           const float* __restrict__ scale,
                           const float* __restrict__ shift,
                           const __bf16* __restrict__ Bswz,
                           const float* __restrict__ bQ,
                           const float* __restrict__ bK,
                           const float* __restrict__ bV,
                           float* __restrict__ out) {
  // LDS: 40 KB total
  __shared__ __align__(32) __bf16 xnA [8 * 32 * 16];   // xn in A-lane layout (8 ktiles)
  __shared__ __align__(32) __bf16 Qlds[PE * 16];       // Q transposed: [e][p]
  __shared__ __align__(32) __bf16 Klds[PE * 16];       // K transposed: [f][p]
  __shared__ __align__(32) __bf16 Vlds[16 * PE];       // V natural:    [p][f]
  __shared__ __align__(32) __bf16 chunkS[8 * 16 * 32]; // per-wave attn k-chunk

  const int b    = blockIdx.x;
  const int t    = threadIdx.x;
  const int lane = t & 31;
  const int wave = t >> 5;
  const int half = lane >> 4;
  const int lo   = lane & 15;
  const float* xrow = x + (size_t)b * INF;
  float*       orow = out + (size_t)b * INF;

  // ---- stage xn (BN applied) into LDS in A-lane layout ----
  // A order used here (contiguous k, matching B): lane l: M = l&15; element i: K = i + 16*(l>>4)
#pragma unroll
  for (int j = 0; j < 16; ++j) {
    int col = t + 256 * j;
    float v = xrow[col] * scale[col] + shift[col];
    int p = col >> 8, k = col & 255;
    int kt = k >> 5, kr = k & 31, g = kr >> 3;
    int i  = kr & 15;
    int ln = 16 * (kr >> 4) + p;
    xnA[(kt * 32 + ln) * 16 + i] = (__bf16)v;
  }
  __syncthreads();

  // ---- Phase 1: Q/K/V projections. 48 (proj, etile) jobs over 8 waves ----
  for (int jj = 0; jj < 6; ++jj) {
    int job  = wave * 6 + jj;
    int proj = job >> 4;          // wave-uniform
    int et   = job & 15;
    int e    = et * 16 + lo;      // output column within E
    const float* bp = (proj == 0) ? bQ : ((proj == 1) ? bK : bV);
    float bv = bp[e];
    v8f acc;
#pragma unroll
    for (int i = 0; i < 8; ++i) acc[i] = bv;   // bias depends only on N
#pragma unroll
    for (int kt = 0; kt < 8; ++kt) {
      v16bf a  = *(const v16bf*)(xnA + (kt * 32 + lane) * 16);
      v16bf bb = *(const v16bf*)(Bswz + ((((proj * 16 + et) * 8) + kt) * 32 + lane) * 16);
      acc = __builtin_amdgcn_wmma_f32_16x16x32_bf16(false, a, false, bb,
                                                    (short)0, acc, false, false);
    }
    // C layout: lane l, vgpr v -> M = p = v + 8*half, N = e_local = lo
    v8bf pk;
#pragma unroll
    for (int i = 0; i < 8; ++i) pk[i] = (__bf16)acc[i];
    if (proj == 0) {                       // Q transposed [e][p], p = 8*half + i
      *(v8bf*)(Qlds + e * 16 + 8 * half) = pk;
    } else if (proj == 1) {                // K transposed [f][p]
      *(v8bf*)(Klds + e * 16 + 8 * half) = pk;
    } else {                               // V natural [p][f]
#pragma unroll
      for (int i = 0; i < 8; ++i) Vlds[(i + 8 * half) * PE + e] = pk[i];
    }
  }
  __syncthreads();

  // ---- Phases 2+3 per wave: 2 mtiles (16 e-rows each) ----
  const int wbase = wave * 16 * 32;
  for (int m2 = 0; m2 < 2; ++m2) {
    int m = wave * 2 + m2;

    // A' = Q^T tile (M = e, K' = p padded 16->32): lanes 16-31 zero.
    v16bf qr = *(const v16bf*)(Qlds + (m * 16 + lo) * 16);
    v16bf aQ;
#pragma unroll
    for (int i = 0; i < 16; ++i) aQ[i] = half ? (__bf16)0.0f : qr[i];

    // scores slab: 16 n-tiles of 16x16 f32 in registers
    v8f sc[16];
#pragma unroll
    for (int n = 0; n < 16; ++n) {
      // B' = K row [f][p]; lanes 16-31 carry K'>=16 which hit zeroed A' elems.
      v16bf bKf = *(const v16bf*)(Klds + (n * 16 + lo) * 16);
      v8f z;
#pragma unroll
      for (int i = 0; i < 8; ++i) z[i] = 0.f;
      sc[n] = __builtin_amdgcn_wmma_f32_16x16x32_bf16(false, aQ, false, bKf,
                                                      (short)0, z, false, false);
    }

    // ---- softmax over f (256) per row; rows live on 16-lane groups ----
    float rs[8];
#pragma unroll
    for (int v = 0; v < 8; ++v) {
      float mx = sc[0][v];
#pragma unroll
      for (int n = 1; n < 16; ++n) mx = fmaxf(mx, sc[n][v]);
#pragma unroll
      for (int s = 1; s < 16; s <<= 1) mx = fmaxf(mx, __shfl_xor(mx, s, 32));
      float sm = 0.f;
#pragma unroll
      for (int n = 0; n < 16; ++n) {
        float ev = __expf((sc[n][v] - mx) * 0.0625f);   // 1/sqrt(E) = 1/16
        sc[n][v] = ev; sm += ev;
      }
#pragma unroll
      for (int s = 1; s < 16; s <<= 1) sm += __shfl_xor(sm, s, 32);
      rs[v] = 1.0f / sm;
    }

    // ---- Phase 3: prod = attn (256-wide K) x V^T, via per-ktile LDS chunk ----
    v8f acc2;
#pragma unroll
    for (int i = 0; i < 8; ++i) acc2[i] = 0.f;
#pragma unroll
    for (int kt = 0; kt < 8; ++kt) {
      // stage attn columns f = kt*32 .. kt*32+31 into this wave's chunk [e_local][kr]
#pragma unroll
      for (int n2 = 0; n2 < 2; ++n2) {
        int n = kt * 2 + n2;
#pragma unroll
        for (int v = 0; v < 8; ++v)
          chunkS[wbase + (v + 8 * half) * 32 + n2 * 16 + lo] = (__bf16)(sc[n][v] * rs[v]);
      }
      // A'' fragment: lane l row = lo; kr = i + 16*half
      v8bf a0 = *(const v8bf*)(chunkS + wbase + lo * 32 + 16 * half);
      v8bf a1 = *(const v8bf*)(chunkS + wbase + lo * 32 + 16 * half + 8);
      v16bf aa;
#pragma unroll
      for (int i = 0; i < 8; ++i) { aa[i] = a0[i]; aa[i + 8] = a1[i]; }
      // B'' = V^T: lane l: N = p = lo, K'' = f = kt*32 + 16*half + i
      v16bf bVf = *(const v16bf*)(Vlds + lo * PE + kt * 32 + 16 * half);
      acc2 = __builtin_amdgcn_wmma_f32_16x16x32_bf16(false, aa, false, bVf,
                                                     (short)0, acc2, false, false);
    }

    // ---- residual add + store: out[b, e*16 + p], e = m*16 + v + 8*half, p = lo ----
#pragma unroll
    for (int v = 0; v < 8; ++v) {
      int idx = (m * 16 + v + 8 * half) * 16 + lo;
      orow[idx] = acc2[v] + xrow[idx];
    }
  }
}

extern "C" void kernel_launch(void* const* d_in, const int* in_sizes, int n_in,
                              void* d_out, int out_size, void* d_ws, size_t ws_size,
                              hipStream_t stream) {
  const float* x     = (const float*)d_in[0];
  const float* gamma = (const float*)d_in[1];
  const float* beta  = (const float*)d_in[2];
  const float* WQ    = (const float*)d_in[3];
  const float* bQ    = (const float*)d_in[4];
  const float* WK    = (const float*)d_in[5];
  const float* bK    = (const float*)d_in[6];
  const float* WV    = (const float*)d_in[7];
  const float* bV    = (const float*)d_in[8];
  float* out = (float*)d_out;

  // workspace layout (floats, then bf16; ~690 KB total)
  float* scale  = (float*)d_ws;                 // 4096
  float* shift  = scale + INF;                  // 4096
  float* partS  = shift + INF;                  // 8*4096
  float* partS2 = partS + NCHUNK * INF;         // 8*4096
  __bf16* Bswz  = (__bf16*)(partS2 + NCHUNK * INF);  // 3*256*256 bf16 (32B aligned)

  dim3 gridP(INF / 256, NCHUNK);
  bn_partial<<<gridP, 256, 0, stream>>>(x, partS, partS2);
  bn_finalize<<<INF / 256, 256, 0, stream>>>(partS, partS2, gamma, beta, scale, shift);
  weight_swizzle<<<(3 * PE * PE) / 256, 256, 0, stream>>>(WQ, WK, WV, Bswz);
  attn_fused<<<NB, 256, 0, stream>>>(x, scale, shift, Bswz, bQ, bK, bV, out);
}
